// DecoderLayer_2190433321077
// MI455X (gfx1250) — matrix operand order repaired and checked by reading, not compiled, not run on hardware
//
#include <hip/hip_runtime.h>
#include <cstdint>
#include <cstddef>

// Problem constants (reference): B=2, S=2048, D=1024, H=16, F=4096, DK=64
#define BB  2
#define SS  2048
#define DD  1024
#define HH  16
#define FF  4096
#define DKK 64
#define MM  (BB * SS) // 4096 rows for all GEMMs

typedef __attribute__((ext_vector_type(16))) __bf16         v16bf;
typedef __attribute__((ext_vector_type(8)))  __bf16         v8bf;
typedef __attribute__((ext_vector_type(8)))  float          v8f;
typedef __attribute__((ext_vector_type(8)))  unsigned short v8u;
typedef __attribute__((ext_vector_type(16))) unsigned short v16u;
typedef __attribute__((ext_vector_type(4)))  unsigned int   v4u32;
typedef __attribute__((ext_vector_type(8)))  int            v8i32;
typedef __attribute__((ext_vector_type(4)))  int            v4i32;

#if defined(__has_builtin)
#if __has_builtin(__builtin_amdgcn_tensor_load_to_lds)
#define HAVE_TDM 1
#endif
#endif

// ---------------------------------------------------------------- helpers ---
__device__ __forceinline__ unsigned short f32_to_bf16(float f) {
  union { float f; unsigned u; } v; v.f = f;
  unsigned r = v.u + 0x7FFFu + ((v.u >> 16) & 1u);   // round-to-nearest-even
  return (unsigned short)(r >> 16);
}

// A-matrix fragment (16x32 bf16, M = lane&15):
//   elements 0..7  -> K = 8*half + e ; elements 8..15 -> K = 16 + 8*half + (e-8)
// => two contiguous 16B LDS loads. (ISA 7.12.2, 16-bit A 16x32 table)
__device__ __forceinline__ v16bf frag_A(const unsigned short* rowbase, int half) {
  v8u lo = *(const v8u*)(rowbase + 8 * half);
  v8u hi = *(const v8u*)(rowbase + 16 + 8 * half);
  v8bf l = __builtin_bit_cast(v8bf, lo);
  v8bf h = __builtin_bit_cast(v8bf, hi);
  return __builtin_shufflevector(l, h, 0,1,2,3,4,5,6,7,8,9,10,11,12,13,14,15);
}

// B-matrix fragment (32x16 bf16, N = lane&15):
//   elements 0..7  -> K = 8*half + e ; elements 8..15 -> K = 16 + 8*half + (e-8)
__device__ __forceinline__ v16bf frag_B(const unsigned short* rowbase, int half) {
  v8u lo = *(const v8u*)(rowbase + 8 * half);
  v8u hi = *(const v8u*)(rowbase + 16 + 8 * half);
  v8bf l = __builtin_bit_cast(v8bf, lo);
  v8bf h = __builtin_bit_cast(v8bf, hi);
  return __builtin_shufflevector(l, h, 0,1,2,3,4,5,6,7,8,9,10,11,12,13,14,15);
}

__device__ __forceinline__ v8f wmma_bf16(v16bf a, v16bf b, v8f c) {
  return __builtin_amdgcn_wmma_f32_16x16x32_bf16(false, a, false, b, (short)0, c,
                                                 false, false);
}

#ifdef HAVE_TDM
// 2D TENSOR_LOAD_TO_LDS: tile_d1 rows of tile_d0 bf16 elements, global row
// stride = stride_elems, LDS padded by pad_amount DWORD-code after every
// pad_interval DWORD-code stored (ISA cdna5 8.3-8.5 D# layout).
// 6-arg builtin form (clang-23 / therock-10.0): groups 2/3 + extra group zero.
__device__ __forceinline__ void tdm_load_2d_bf16(unsigned lds_addr, const void* gaddr,
                                                 unsigned tile_d0, unsigned tile_d1,
                                                 unsigned tensor_d0, unsigned tensor_d1,
                                                 unsigned stride_elems,
                                                 unsigned pad_interval,
                                                 unsigned pad_amount) {
  const unsigned long long ga = (unsigned long long)(uintptr_t)gaddr;
  v4u32 g0;
  g0[0] = 1u;                                   // count=1 (user descriptor)
  g0[1] = lds_addr;                             // lds_addr [63:32]
  g0[2] = (unsigned)(ga & 0xFFFFFFFFu);         // global_addr [95:64]
  g0[3] = (unsigned)((ga >> 32) & 0x01FFFFFFu)  // global_addr [120:96]
        | (2u << 30);                           // type=2 ("image") [127:126]
  v8i32 g1;
  g1[0] = (int)((1u << 16)                      // data_size = 2 bytes
              | (1u << 20)                      // pad_enable
              | (pad_interval << 22)            // pad_interval [24:22]
              | (pad_amount << 25));            // pad_amount   [31:25]
  g1[1] = (int)((tensor_d0 & 0xFFFFu) << 16);   // tensor_dim0[15:0] @ [63:48]
  g1[2] = (int)((tensor_d0 >> 16) | ((tensor_d1 & 0xFFFFu) << 16));
  g1[3] = (int)((tensor_d1 >> 16) | (tile_d0 << 16));   // tile_dim0 @ [127:112]
  g1[4] = (int)(tile_d1 & 0xFFFFu);             // tile_dim1; tile_dim2=0 (2D)
  g1[5] = (int)stride_elems;                    // tensor_dim0_stride low 32
  g1[6] = 0;
  g1[7] = 0;
  const v4i32 z4 = {0, 0, 0, 0};                // groups 2/3 unused for 2D tiles
  const v8i32 z8 = {0, 0, 0, 0, 0, 0, 0, 0};
  __builtin_amdgcn_tensor_load_to_lds(g0, g1, z4, z4, z8, 0);
}
#endif

// -------------------------------------------------------------- cvt kernels -
// weights: f32 [K][N] -> bf16 transposed [N][K] (TDM-friendly: contiguous K rows)
__global__ __launch_bounds__(256)
void cvt_bf16_t_kernel(const float* __restrict__ in, unsigned short* __restrict__ out,
                       int N, int kshift) {
  const int K = 1 << kshift;
  const int i = blockIdx.x * 256 + threadIdx.x;   // output index over N*K
  const int n = i >> kshift, k = i & (K - 1);
  out[i] = f32_to_bf16(in[(size_t)k * N + n]);
}

// ------------------------------------------------------------- GEMM kernel --
// C[M=4096, N] = act(A[4096, K] @ W[K, N] + bias); A f32 (cvt->bf16 in LDS),
// weights given transposed bf16 Wt[N][K]; f32 accumulate/output.
// Block tile 128x128, K-step 32.  8 waves: 4 along M x 2 along N.
// Wt tile staged by the Tensor Data Mover (wave 0), A tile by VALU conversion.
template <int RELU>
__global__ __launch_bounds__(256)
void gemm_bias_kernel(const float* __restrict__ A, const unsigned short* __restrict__ Wt,
                      const float* __restrict__ bias, float* __restrict__ C,
                      int N, int K) {
  __shared__ unsigned short lA[128 * 40];   // [row][k], pad 32->40 elems
  __shared__ unsigned short lBt[128 * 40];  // [n][k], pad 32->40 elems

  const int tid    = threadIdx.x;
  const int lane16 = tid & 15;
  const int half   = (tid >> 4) & 1;
  const int wid    = tid >> 5;
  const int waveM  = wid & 3;
  const int waveN  = wid >> 2;
  const int bm = blockIdx.y * 128, bn = blockIdx.x * 128;
#ifdef HAVE_TDM
  const unsigned ldsB = (unsigned)(uintptr_t)(void*)&lBt[0];
#endif

  v8f acc[2][4];
#pragma unroll
  for (int i = 0; i < 2; ++i)
#pragma unroll
    for (int j = 0; j < 4; ++j) acc[i][j] = (v8f){0,0,0,0,0,0,0,0};

  for (int kt = 0; kt < K; kt += 32) {
    __syncthreads();
#ifdef HAVE_TDM
    if (wid == 0) {
      // 128 rows x 32 bf16 (64B) per row; pad 16B/row -> LDS stride 40 ushorts.
      tdm_load_2d_bf16(ldsB, Wt + (size_t)bn * K + kt,
                       /*tile_d0=*/32, /*tile_d1=*/128,
                       /*tensor_d0=*/(unsigned)K, /*tensor_d1=*/128,
                       /*stride=*/(unsigned)K,
                       /*pad_interval=*/3 /*16 DW = 64B*/,
                       /*pad_amount=*/3 /*4 DW = 16B*/);
    }
#endif
    { // stage A tile 128x32 f32 -> bf16 (all 256 threads, overlaps the DMA)
      const int cg = tid & 7;
      const int rg = tid >> 3;
#pragma unroll
      for (int i = 0; i < 4; ++i) {
        const int r = rg * 4 + i;
        const float4 f = *(const float4*)(A + (size_t)(bm + r) * K + kt + cg * 4);
        unsigned short* d = &lA[r * 40 + cg * 4];
        d[0] = f32_to_bf16(f.x); d[1] = f32_to_bf16(f.y);
        d[2] = f32_to_bf16(f.z); d[3] = f32_to_bf16(f.w);
      }
    }
#ifdef HAVE_TDM
    if (wid == 0) __builtin_amdgcn_s_wait_tensorcnt(0);
#else
    { // fallback: manual staging from Wt[N][K] (already transposed in global)
      const int nrow = tid >> 1;            // 0..127
      const int kh   = (tid & 1) * 16;      // 0 or 16
      const v16u w = *(const v16u*)(Wt + (size_t)(bn + nrow) * K + kt + kh);
#pragma unroll
      for (int j = 0; j < 16; ++j) lBt[nrow * 40 + kh + j] = w[j];
    }
#endif
    __syncthreads();

    v16bf af[2], bfv[4];
#pragma unroll
    for (int mt = 0; mt < 2; ++mt)
      af[mt] = frag_A(&lA[(waveM * 32 + mt * 16 + lane16) * 40], half);
#pragma unroll
    for (int nt = 0; nt < 4; ++nt)
      bfv[nt] = frag_B(&lBt[(waveN * 64 + nt * 16 + lane16) * 40], half);
#pragma unroll
    for (int mt = 0; mt < 2; ++mt)
#pragma unroll
      for (int nt = 0; nt < 4; ++nt)
        acc[mt][nt] = wmma_bf16(af[mt], bfv[nt], acc[mt][nt]);
  }

  // epilogue: C/D layout -> row = r + 8*half, col = lane16
#pragma unroll
  for (int mt = 0; mt < 2; ++mt) {
    const int row0 = bm + waveM * 32 + mt * 16 + 8 * half;
#pragma unroll
    for (int nt = 0; nt < 4; ++nt) {
      const int col = bn + waveN * 64 + nt * 16 + lane16;
      const float bv = bias[col];
#pragma unroll
      for (int r = 0; r < 8; ++r) {
        float v = acc[mt][nt][r] + bv;
        if (RELU) v = fmaxf(v, 0.0f);
        C[(size_t)(row0 + r) * N + col] = v;
      }
    }
  }
}

// ------------------------------------------------------- flash attention ----
// Q,K,V,O: f32 [B, S, D] with head h in columns [h*64, h*64+64).
template <int CAUSAL>
__global__ __launch_bounds__(256)
void attn_kernel(const float* __restrict__ Q, const float* __restrict__ Kk,
                 const float* __restrict__ Vv, float* __restrict__ O, int SK) {
  __shared__ unsigned short lK[32 * 72];       // [key][dk]   (64 -> 72 pad)
  __shared__ unsigned short lVt[64 * 40];      // [dk][key]   (32 -> 40 pad)
  __shared__ unsigned short lP[8 * 16 * 40];   // per-wave P staging [q][key]

  const int tid    = threadIdx.x;
  const int lane16 = tid & 15;
  const int half   = (tid >> 4) & 1;
  const int wid    = tid >> 5;
  const int b = blockIdx.z, h = blockIdx.y;
  const int q0 = blockIdx.x * 128 + wid * 16;
  const size_t qb  = (size_t)b * SS * DD;
  const size_t kvb = (size_t)b * (size_t)SK * DD;
  const int hoff = h * DKK;

  v16bf qf[2];
  {
    union { v16bf v; unsigned short u[16]; } t;
#pragma unroll
    for (int ks = 0; ks < 2; ++ks) {
      const float* qp = Q + qb + (size_t)(q0 + lane16) * DD + hoff + ks * 32 + 8 * half;
#pragma unroll
      for (int e = 0; e < 8; ++e) t.u[e] = f32_to_bf16(qp[e]);
#pragma unroll
      for (int e = 0; e < 8; ++e) t.u[8 + e] = f32_to_bf16(qp[16 + e]);
      qf[ks] = t.v;
    }
  }

  v8f oacc[4];
#pragma unroll
  for (int i = 0; i < 4; ++i) oacc[i] = (v8f){0,0,0,0,0,0,0,0};
  float m[8], l[8];
#pragma unroll
  for (int r = 0; r < 8; ++r) { m[r] = -1e30f; l[r] = 0.0f; }

  const int nkb = CAUSAL ? (blockIdx.x * 128 + 128) / 32 : (SK / 32);
  for (int kbi = 0; kbi < nkb; ++kbi) {
    __syncthreads();
    { // cooperative stage: K [32][64] row-major, V transposed [64][32]
      const int cg = tid & 7, kr = tid >> 3;
      const float* kp = Kk + kvb + (size_t)(kbi * 32 + kr) * DD + hoff + cg * 8;
      const float4 a = *(const float4*)kp;
      const float4 c = *(const float4*)(kp + 4);
      unsigned short* d = &lK[kr * 72 + cg * 8];
      d[0]=f32_to_bf16(a.x); d[1]=f32_to_bf16(a.y); d[2]=f32_to_bf16(a.z); d[3]=f32_to_bf16(a.w);
      d[4]=f32_to_bf16(c.x); d[5]=f32_to_bf16(c.y); d[6]=f32_to_bf16(c.z); d[7]=f32_to_bf16(c.w);
      const float* vp = Vv + kvb + (size_t)(kbi * 32 + kr) * DD + hoff + cg * 8;
      const float4 e4 = *(const float4*)vp;
      const float4 f4 = *(const float4*)(vp + 4);
      const float vvv[8] = {e4.x, e4.y, e4.z, e4.w, f4.x, f4.y, f4.z, f4.w};
#pragma unroll
      for (int j = 0; j < 8; ++j) lVt[(cg * 8 + j) * 40 + kr] = f32_to_bf16(vvv[j]);
    }
    __syncthreads();
    if (CAUSAL && kbi * 32 > q0 + 15) continue;  // fully-masked block (wave-uniform)

    v8f sc[2];
#pragma unroll
    for (int nt = 0; nt < 2; ++nt) {
      v8f s = (v8f){0,0,0,0,0,0,0,0};
      s = wmma_bf16(qf[0], frag_B(&lK[(nt * 16 + lane16) * 72 + 0],  half), s);
      s = wmma_bf16(qf[1], frag_B(&lK[(nt * 16 + lane16) * 72 + 32], half), s);
      sc[nt] = s * 0.125f;
    }
    if (CAUSAL) {
#pragma unroll
      for (int nt = 0; nt < 2; ++nt)
#pragma unroll
        for (int r = 0; r < 8; ++r) {
          const int key = kbi * 32 + nt * 16 + lane16;
          const int q   = q0 + r + 8 * half;
          if (key > q) sc[nt][r] = -1e9f;
        }
    }

    unsigned short* pst = &lP[wid * 16 * 40];
#pragma unroll
    for (int r = 0; r < 8; ++r) {
      float tmax = fmaxf(sc[0][r], sc[1][r]);
#pragma unroll
      for (int off = 8; off >= 1; off >>= 1) tmax = fmaxf(tmax, __shfl_xor(tmax, off, 32));
      const float mnew = fmaxf(m[r], tmax);
      const float corr = __expf(m[r] - mnew);
      const float p0 = __expf(sc[0][r] - mnew);
      const float p1 = __expf(sc[1][r] - mnew);
      float rs = p0 + p1;
#pragma unroll
      for (int off = 8; off >= 1; off >>= 1) rs += __shfl_xor(rs, off, 32);
      l[r] = l[r] * corr + rs;
      m[r] = mnew;
#pragma unroll
      for (int t2 = 0; t2 < 4; ++t2) oacc[t2][r] *= corr;
      const int prow = r + 8 * half;
      pst[prow * 40 + lane16]      = f32_to_bf16(p0);
      pst[prow * 40 + 16 + lane16] = f32_to_bf16(p1);
    }

    const v16bf pf = frag_A(&pst[lane16 * 40], half);
#pragma unroll
    for (int t2 = 0; t2 < 4; ++t2)
      oacc[t2] = wmma_bf16(pf, frag_B(&lVt[(t2 * 16 + lane16) * 40], half), oacc[t2]);
  }

#pragma unroll
  for (int r = 0; r < 8; ++r) {
    const float inv = 1.0f / l[r];
    const int q = q0 + r + 8 * half;
    float* op = O + qb + (size_t)q * DD + hoff;
#pragma unroll
    for (int t2 = 0; t2 < 4; ++t2) op[t2 * 16 + lane16] = oacc[t2][r] * inv;
  }
}

// ----------------------------------------------------- add + LayerNorm ------
__device__ __forceinline__ float block_sum(float v, float* red) {
  const int tid = threadIdx.x;
#pragma unroll
  for (int off = 16; off >= 1; off >>= 1) v += __shfl_xor(v, off, 32);
  __syncthreads();
  if ((tid & 31) == 0) red[tid >> 5] = v;
  __syncthreads();
  float t = red[0];
#pragma unroll
  for (int i = 1; i < 8; ++i) t += red[i];
  return t;
}

__global__ __launch_bounds__(256)
void add_ln_kernel(const float* __restrict__ X, const float* __restrict__ R,
                   const float* __restrict__ g, const float* __restrict__ be,
                   float* __restrict__ out) {
  __shared__ float red[8];
  const int row = blockIdx.x, tid = threadIdx.x;
  const float* xp = X + (size_t)row * DD;
  const float* rp = R + (size_t)row * DD;
  float v[4];
  float s = 0.0f;
#pragma unroll
  for (int i = 0; i < 4; ++i) { v[i] = xp[tid + i * 256] + rp[tid + i * 256]; s += v[i]; }
  const float mn = block_sum(s, red) * (1.0f / DD);
  float ss = 0.0f;
#pragma unroll
  for (int i = 0; i < 4; ++i) { const float d = v[i] - mn; ss += d * d; }
  const float sd  = sqrtf(block_sum(ss, red) * (1.0f / (DD - 1)));  // ddof=1
  const float inv = 1.0f / (sd + 1e-5f);                            // eps on std
#pragma unroll
  for (int i = 0; i < 4; ++i) {
    const int c = tid + i * 256;
    out[(size_t)row * DD + c] = (v[i] - mn) * inv * g[c] + be[c];
  }
}

// ------------------------------------------------------------- launcher -----
extern "C" void kernel_launch(void* const* d_in, const int* in_sizes, int n_in,
                              void* d_out, int out_size, void* d_ws, size_t ws_size,
                              hipStream_t stream) {
  const float* X   = (const float*)d_in[0];
  const float* y   = (const float*)d_in[1];
  const float* wq1 = (const float*)d_in[2];  const float* bq1 = (const float*)d_in[3];
  const float* wk1 = (const float*)d_in[4];  const float* bk1 = (const float*)d_in[5];
  const float* wv1 = (const float*)d_in[6];  const float* bv1 = (const float*)d_in[7];
  const float* wo1 = (const float*)d_in[8];  const float* bo1 = (const float*)d_in[9];
  const float* wq2 = (const float*)d_in[10]; const float* bq2 = (const float*)d_in[11];
  const float* wk2 = (const float*)d_in[12]; const float* bk2 = (const float*)d_in[13];
  const float* wv2 = (const float*)d_in[14]; const float* bv2 = (const float*)d_in[15];
  const float* wo2 = (const float*)d_in[16]; const float* bo2 = (const float*)d_in[17];
  const float* wf1 = (const float*)d_in[18]; const float* bf1 = (const float*)d_in[19];
  const float* wf2 = (const float*)d_in[20]; const float* bf2 = (const float*)d_in[21];
  const float* g1  = (const float*)d_in[22]; const float* be1 = (const float*)d_in[23];
  const float* g2  = (const float*)d_in[24]; const float* be2 = (const float*)d_in[25];
  const float* g3  = (const float*)d_in[26]; const float* be3 = (const float*)d_in[27];

  // workspace (~144 MB):
  //   [0, 32MB): bf16 transposed weights (8 x D*D + 2 x D*F = 16M elems)
  //   then 7 x (B*S*D) f32 slots: q, k, v, att, o, h1, h2
  //   FFN hidden (B*S*F f32) overlays slots q..att
  const size_t DD2 = (size_t)DD * DD;  // 1M
  const size_t DF  = (size_t)DD * FF;  // 4M
  unsigned short* wb = (unsigned short*)d_ws;
  unsigned short* wq1b = wb + 0 * DD2;
  unsigned short* wk1b = wb + 1 * DD2;
  unsigned short* wv1b = wb + 2 * DD2;
  unsigned short* wo1b = wb + 3 * DD2;
  unsigned short* wq2b = wb + 4 * DD2;
  unsigned short* wk2b = wb + 5 * DD2;
  unsigned short* wv2b = wb + 6 * DD2;
  unsigned short* wo2b = wb + 7 * DD2;
  unsigned short* wf1b = wb + 8 * DD2;
  unsigned short* wf2b = wb + 8 * DD2 + DF;

  float* fa = (float*)((char*)d_ws + (8 * DD2 + 2 * DF) * sizeof(unsigned short));
  const size_t NSD = (size_t)MM * DD;  // 4M floats
  float* s_q   = fa + 0 * NSD;
  float* s_k   = fa + 1 * NSD;
  float* s_v   = fa + 2 * NSD;
  float* s_att = fa + 3 * NSD;
  float* s_o   = fa + 4 * NSD;
  float* h1    = fa + 5 * NSD;
  float* h2    = fa + 6 * NSD;
  float* s_f   = fa;  // overlays q..att

  const dim3 blk(256);
  // weights f32 [K][N] -> bf16 transposed [N][K]
  auto cvt_t = [&](const float* src, unsigned short* dst, size_t total, int N, int kshift) {
    cvt_bf16_t_kernel<<<dim3((unsigned)(total / 256)), blk, 0, stream>>>(src, dst, N, kshift);
  };
  cvt_t(wq1, wq1b, DD2, DD, 10); cvt_t(wk1, wk1b, DD2, DD, 10);
  cvt_t(wv1, wv1b, DD2, DD, 10); cvt_t(wo1, wo1b, DD2, DD, 10);
  cvt_t(wq2, wq2b, DD2, DD, 10); cvt_t(wk2, wk2b, DD2, DD, 10);
  cvt_t(wv2, wv2b, DD2, DD, 10); cvt_t(wo2, wo2b, DD2, DD, 10);
  cvt_t(wf1, wf1b, DF, FF, 10);   // [D][F] -> [F][D], K=D (kshift 10)
  cvt_t(wf2, wf2b, DF, DD, 12);   // [F][D] -> [D][F], K=F (kshift 12)

  const dim3 gD(DD / 128, MM / 128);  // (8, 32)
  const dim3 gF(FF / 128, MM / 128);  // (32, 32)
  const dim3 gA(SS / 128, HH, BB);    // (16, 16, 2)

  // --- masked self-attention on y ---
  gemm_bias_kernel<0><<<gD, blk, 0, stream>>>(y, wq1b, bq1, s_q, DD, DD);
  gemm_bias_kernel<0><<<gD, blk, 0, stream>>>(y, wk1b, bk1, s_k, DD, DD);
  gemm_bias_kernel<0><<<gD, blk, 0, stream>>>(y, wv1b, bv1, s_v, DD, DD);
  attn_kernel<1><<<gA, blk, 0, stream>>>(s_q, s_k, s_v, s_att, SS);
  gemm_bias_kernel<0><<<gD, blk, 0, stream>>>(s_att, wo1b, bo1, s_o, DD, DD);
  add_ln_kernel<<<dim3(MM), blk, 0, stream>>>(s_o, y, g1, be1, h1);

  // --- cross-attention (queries h1, keys/values X) ---
  gemm_bias_kernel<0><<<gD, blk, 0, stream>>>(h1, wq2b, bq2, s_q, DD, DD);
  gemm_bias_kernel<0><<<gD, blk, 0, stream>>>(X,  wk2b, bk2, s_k, DD, DD);
  gemm_bias_kernel<0><<<gD, blk, 0, stream>>>(X,  wv2b, bv2, s_v, DD, DD);
  attn_kernel<0><<<gA, blk, 0, stream>>>(s_q, s_k, s_v, s_att, SS);
  gemm_bias_kernel<0><<<gD, blk, 0, stream>>>(s_att, wo2b, bo2, s_o, DD, DD);
  add_ln_kernel<<<dim3(MM), blk, 0, stream>>>(s_o, h1, g2, be2, h2);

  // --- feed-forward ---
  gemm_bias_kernel<1><<<gF, blk, 0, stream>>>(h2, wf1b, bf1, s_f, FF, DD);
  gemm_bias_kernel<0><<<gD, blk, 0, stream>>>(s_f, wf2b, bf2, s_o, DD, FF);
  add_ln_kernel<<<dim3(MM), blk, 0, stream>>>(s_o, h2, g3, be3, (float*)d_out);
}
